// Reservoir_34041910788220
// MI455X (gfx1250) — matrix operand order repaired and checked by reading, not compiled, not run on hardware
//
#include <hip/hip_runtime.h>
#include <hip/hip_bf16.h>
#include <math.h>

// ---------------------------------------------------------------------------
// Reservoir (echo-state) rollout on MI455X / gfx1250.
//   x_{t} = tanh(u_t @ W_ih^T + b_ih + x_{t-1} @ W_hh^T),  out[(b*H+j)][t] = x_t[b][j]
// SEQ=1024, B=64, IN=512, H=2048.
//
// Design: one persistent kernel, 64 workgroups; WG g owns output units
// j in [32g, 32g+32). Its 32x2048 slice of W_hh (bf16, 128KB) and 32x512
// slice of W_ih (32KB) live in LDS for the whole rollout (CDNA5: 320KB/WGP).
// Per timestep each of the 8 waves does one 16x16 tile via
// v_wmma_f32_16x16x32_bf16 over K=512 (input) + K=2048 (recurrent),
// then tanh + store, then a grid-wide generation barrier.
// ---------------------------------------------------------------------------

#define SEQL   1024
#define BATCH  64
#define INSZ   512
#define HSZ    2048
#define NBLK   64          // HSZ / JPB
#define JPB    32          // output units per workgroup
#define NTHR   256         // 8 wave32s

typedef __attribute__((ext_vector_type(16))) __bf16 v16bf;
typedef __attribute__((ext_vector_type(8)))  float  v8f;

union FragU {
    uint4 u[2];   // two 16B halves
    v16bf v;      // 16 bf16 = 8 VGPRs
};

// ---------------------------------------------------------------------------
// Prep: fp32 input -> bf16 workspace copy; zero both state buffers; reset
// the grid barrier. Re-run every launch so graph replay is deterministic.
// ---------------------------------------------------------------------------
__global__ void reservoir_prep(const float* __restrict__ inp,
                               __bf16* __restrict__ in_bf,
                               __bf16* __restrict__ xbufs,   // 2 * BATCH*HSZ
                               unsigned* __restrict__ bar,
                               long n_in, long n_x)
{
    long i      = (long)blockIdx.x * blockDim.x + threadIdx.x;
    long stride = (long)gridDim.x * blockDim.x;
    for (long k = i; k < n_in; k += stride)
        in_bf[k] = (__bf16)inp[k];
    for (long k = i; k < n_x; k += stride)
        xbufs[k] = (__bf16)0.0f;
    if (i == 0) { bar[0] = 0u; bar[1] = 0u; }
}

// ---------------------------------------------------------------------------
// Grid-wide generation barrier (all NBLK blocks are co-resident: 64 WGs,
// 160KB LDS each, on a part with >=64 WGPs).
// ---------------------------------------------------------------------------
__device__ __forceinline__ void grid_sync(unsigned* bar, unsigned nblk)
{
    __syncthreads();
    if (threadIdx.x == 0) {
        __threadfence();  // make this block's stores device-visible
        unsigned gen = __hip_atomic_load(&bar[1], __ATOMIC_RELAXED,
                                         __HIP_MEMORY_SCOPE_AGENT);
        unsigned old = __hip_atomic_fetch_add(&bar[0], 1u, __ATOMIC_ACQ_REL,
                                              __HIP_MEMORY_SCOPE_AGENT);
        if (old == nblk - 1u) {
            __hip_atomic_store(&bar[0], 0u, __ATOMIC_RELAXED,
                               __HIP_MEMORY_SCOPE_AGENT);
            __hip_atomic_store(&bar[1], gen + 1u, __ATOMIC_RELEASE,
                               __HIP_MEMORY_SCOPE_AGENT);
        } else {
            while (__hip_atomic_load(&bar[1], __ATOMIC_ACQUIRE,
                                     __HIP_MEMORY_SCOPE_AGENT) == gen)
                __builtin_amdgcn_s_sleep(2);
        }
    }
    __syncthreads();
}

// ---------------------------------------------------------------------------
// Persistent rollout kernel.
// ---------------------------------------------------------------------------
__global__ __launch_bounds__(NTHR, 1)
void reservoir_rollout(const __bf16* __restrict__ u_bf,   // [SEQ][B][IN] bf16
                       const float*  __restrict__ w_ih,   // [H][IN] fp32
                       const float*  __restrict__ b_ih,   // [H] fp32
                       const float*  __restrict__ w_hh,   // [H][H] fp32
                       float*        __restrict__ out,    // [B*H][SEQ] fp32
                       __bf16*       __restrict__ xbuf0,  // [B][H] bf16
                       __bf16*       __restrict__ xbuf1,  // [B][H] bf16
                       unsigned*     __restrict__ bar)
{
    // 128KB + 32KB = 160KB of LDS: whole recurrent+input weight slice resident.
    __shared__ __bf16 lds_whh[JPB][HSZ];
    __shared__ __bf16 lds_wih[JPB][INSZ];

    const int tid = threadIdx.x;
    const int j0  = blockIdx.x * JPB;

    // One-time: convert this WG's weight slices fp32 -> bf16 into LDS.
    for (int idx = tid; idx < JPB * HSZ; idx += NTHR) {
        int lj = idx / HSZ, k = idx % HSZ;
        lds_whh[lj][k] = (__bf16)w_hh[(size_t)(j0 + lj) * HSZ + k];
    }
    for (int idx = tid; idx < JPB * INSZ; idx += NTHR) {
        int lj = idx / INSZ, k = idx % INSZ;
        lds_wih[lj][k] = (__bf16)w_ih[(size_t)(j0 + lj) * INSZ + k];
    }
    __syncthreads();

    // Wave -> tile mapping: 8 waves = 4 batch-row tiles x 2 column tiles.
    const int lane     = tid & 31;
    const int wave     = tid >> 5;
    const int rowBase  = (wave & 3) * 16;        // batch rows [rowBase, rowBase+16)
    const int colLocal = (wave >> 2) * 16;       // local cols [colLocal, colLocal+16)
    const int n        = lane & 15;              // A: row M; B/D: column N
    const int hiHalf   = lane >> 4;              // lane group 0/1
    const int jcol     = j0 + colLocal + n;      // global output unit of this lane
    const float bias   = b_ih[jcol];

    const __bf16* wihp = &lds_wih[colLocal + n][0];
    const __bf16* whhp = &lds_whh[colLocal + n][0];

    for (int t = 0; t < SEQL; ++t) {
        const __bf16* xr = (t & 1) ? xbuf1 : xbuf0;   // read state
        __bf16*       xw = (t & 1) ? xbuf0 : xbuf1;   // write state

        v8f acc;
        #pragma unroll
        for (int e = 0; e < 8; ++e) acc[e] = bias;

        // ---- input GEMM: u_t (16x512) x W_ih-slice (512x16) ----
        const __bf16* ubase =
            u_bf + (size_t)t * (BATCH * INSZ) + (size_t)(rowBase + n) * INSZ;
        __builtin_prefetch(ubase + BATCH * INSZ, 0, 1);  // next timestep's rows
        #pragma unroll 4
        for (int k0 = 0; k0 < INSZ; k0 += 32) {
            FragU a, b;
            // A (16-bit 16x32 layout): lanes 0-15 hold K 0-7 / 16-23,
            // lanes 16-31 hold K 8-15 / 24-31 -> two 16B contiguous loads.
            a.u[0] = *(const uint4*)(ubase + k0 + hiHalf * 8);
            a.u[1] = *(const uint4*)(ubase + k0 + hiHalf * 8 + 16);
            // B (16-bit 32x16 layout): lane = column, lanes 0-15 hold K 0-7 / 16-23,
            // lanes 16-31 hold K 8-15 / 24-31 -> two 16B loads from LDS.
            b.u[0] = *(const uint4*)(wihp + k0 + hiHalf * 8);
            b.u[1] = *(const uint4*)(wihp + k0 + hiHalf * 8 + 16);
            acc = __builtin_amdgcn_wmma_f32_16x16x32_bf16(
                      false, a.v, false, b.v, (short)0, acc, false, false);
        }

        // ---- recurrent GEMM: x (16x2048) x W_hh-slice (2048x16) ----
        const __bf16* xbase = xr + (size_t)(rowBase + n) * HSZ;
        #pragma unroll 4
        for (int k0 = 0; k0 < HSZ; k0 += 32) {
            FragU a, b;
            a.u[0] = *(const uint4*)(xbase + k0 + hiHalf * 8);
            a.u[1] = *(const uint4*)(xbase + k0 + hiHalf * 8 + 16);
            b.u[0] = *(const uint4*)(whhp + k0 + hiHalf * 8);
            b.u[1] = *(const uint4*)(whhp + k0 + hiHalf * 8 + 16);
            acc = __builtin_amdgcn_wmma_f32_16x16x32_bf16(
                      false, a.v, false, b.v, (short)0, acc, false, false);
        }

        // ---- tanh, emit output column t, publish next state ----
        // D layout: lane L holds N = L%16; VGPR g holds M = g + 8*(L/16).
        #pragma unroll
        for (int g = 0; g < 8; ++g) {
            float v   = tanhf(acc[g]);
            int  brow = rowBase + g + 8 * hiHalf;
            out[((size_t)brow * HSZ + jcol) * SEQL + (size_t)t] = v;
            xw[(size_t)brow * HSZ + jcol] = (__bf16)v;
        }

        grid_sync(bar, NBLK);
    }
}

// ---------------------------------------------------------------------------
// Launch: d_in order = {input, weight_ih, bias_ih, weight_hh}
// Workspace: [in_bf16: SEQ*B*IN*2][xbuf0: B*H*2][xbuf1: B*H*2][barrier]
//   = 64MiB + 256KiB + 256KiB + 8B  (~67.6 MB)
// ---------------------------------------------------------------------------
extern "C" void kernel_launch(void* const* d_in, const int* in_sizes, int n_in,
                              void* d_out, int out_size, void* d_ws, size_t ws_size,
                              hipStream_t stream)
{
    const float* inp  = (const float*)d_in[0];
    const float* w_ih = (const float*)d_in[1];
    const float* b_ih = (const float*)d_in[2];
    const float* w_hh = (const float*)d_in[3];

    char* ws = (char*)d_ws;
    const size_t in_bf_bytes = (size_t)SEQL * BATCH * INSZ * 2;
    const size_t x_bytes     = (size_t)BATCH * HSZ * 2;

    __bf16*   in_bf = (__bf16*)ws;
    __bf16*   xbuf0 = (__bf16*)(ws + in_bf_bytes);
    __bf16*   xbuf1 = (__bf16*)(ws + in_bf_bytes + x_bytes);
    unsigned* bar   = (unsigned*)(ws + in_bf_bytes + 2 * x_bytes);

    reservoir_prep<<<1024, 256, 0, stream>>>(
        inp, in_bf, xbuf0, bar,
        (long)SEQL * BATCH * INSZ, (long)2 * BATCH * HSZ);

    reservoir_rollout<<<NBLK, NTHR, 0, stream>>>(
        in_bf, w_ih, b_ih, w_hh, (float*)d_out, xbuf0, xbuf1, bar);
}
